// GAT1_22411139350776
// MI455X (gfx1250) — matrix operand order repaired and checked by reading, not compiled, not run on hardware
//
#include <hip/hip_runtime.h>
#include <cstdint>
#include <cstddef>

// ---------------------------------------------------------------------------
// GAT forward for MI455X (gfx1250, wave32, WMMA + TDM).
//   h   = x @ W_fc                       (bf16 WMMA, f32 acc)
//   P   = adj ? exp(a2[j]-M) : 0         (a1 cancels in row softmax)
//   out = (P @ h_pad) / (P @ ones)       (ones folded in as column 128)
// adj (256 MB) streamed once -> HBM-bound ~11us floor. B tiles of the
// 17 GFLOP attention GEMM are staged block-wide into LDS by the Tensor
// Data Mover (double-buffered), consumed via bank-conflict-free ds_load.
// ---------------------------------------------------------------------------

typedef __attribute__((ext_vector_type(16))) __bf16 v16bf;
typedef __attribute__((ext_vector_type(8)))  __bf16 v8bf;
typedef __attribute__((ext_vector_type(4)))  __bf16 v4bf;
typedef __attribute__((ext_vector_type(8)))  float  v8f;
typedef __attribute__((ext_vector_type(4)))  unsigned int u32x4;
typedef __attribute__((ext_vector_type(8)))  int i32x8;
typedef __attribute__((ext_vector_type(4)))  int i32x4;

#define NN    8192
#define DIN   512
#define DOUT  128
#define HTR   144          // ht rows: 128 real + ones row (128) + zero pad
#define ACOL  144          // partial accumulator columns (col 128 = Z)
#define KSEG  4
#define KSEGL (NN / KSEG)  // 2048
#define TROW  36           // LDS tile row stride in elements (64B + 8B TDM pad)

// ---------------- prep kernels ----------------

__global__ void k_cvt_x(const float* __restrict__ x, __bf16* __restrict__ xbf, int n) {
  int i = blockIdx.x * blockDim.x + threadIdx.x;
  int st = gridDim.x * blockDim.x;
  for (; i < n; i += st) xbf[i] = (__bf16)x[i];
}

// wt[n][k] = W_fc[k][n]  (bf16, row-major [128][512]) -> contiguous B loads
__global__ void k_cvt_w(const float* __restrict__ W, __bf16* __restrict__ wt) {
  int idx = blockIdx.x * blockDim.x + threadIdx.x;   // 128*512
  int n = idx >> 9, k = idx & 511;
  wt[idx] = (__bf16)W[k * DOUT + n];
}

// ht rows 128..143: row 128 = ones (Z column), rest zero
__global__ void k_init_htpad(__bf16* __restrict__ htp) {
  int idx = blockIdx.x * blockDim.x + threadIdx.x;   // 16*8192
  int r = idx >> 13;
  int j = idx & (NN - 1);
  htp[(size_t)(128 + r) * NN + j] = (r == 0) ? (__bf16)1.0f : (__bf16)0.0f;
}

// ---------------- h = x @ W_fc  (bf16 WMMA) ----------------
__global__ void k_fc(const __bf16* __restrict__ xbf, const __bf16* __restrict__ wt,
                     float* __restrict__ h, __bf16* __restrict__ htp) {
  const int lane = threadIdx.x & 31;
  const int wave = blockIdx.x * (blockDim.x >> 5) + (threadIdx.x >> 5); // 0..511
  const int lo = lane & 15, hi = lane >> 4;
  const int i0 = wave * 16;

  v8f acc[8];
#pragma unroll
  for (int t = 0; t < 8; ++t)
#pragma unroll
    for (int e = 0; e < 8; ++e) acc[t][e] = 0.0f;

  const __bf16* arow = xbf + (size_t)(i0 + lo) * DIN;
  for (int k0 = 0; k0 < DIN; k0 += 32) {
    v8bf a0 = *(const v8bf*)(arow + k0 + hi * 8);
    v8bf a1 = *(const v8bf*)(arow + k0 + 16 + hi * 8);
    v16bf A;
#pragma unroll
    for (int e = 0; e < 8; ++e) { A[e] = a0[e]; A[e + 8] = a1[e]; }
    v16bf Bt[8];
#pragma unroll
    for (int nt = 0; nt < 8; ++nt) {
      const __bf16* brow = wt + (size_t)(nt * 16 + lo) * DIN + k0 + hi * 8;
      v8bf b0 = *(const v8bf*)(brow);
      v8bf b1 = *(const v8bf*)(brow + 16);
#pragma unroll
      for (int e = 0; e < 8; ++e) { Bt[nt][e] = b0[e]; Bt[nt][e + 8] = b1[e]; }
    }
#pragma unroll
    for (int nt = 0; nt < 8; ++nt)
      acc[nt] = __builtin_amdgcn_wmma_f32_16x16x32_bf16(
          false, A, false, Bt[nt], (short)0, acc[nt], false, false);
  }
#pragma unroll
  for (int nt = 0; nt < 8; ++nt)
#pragma unroll
    for (int r = 0; r < 8; ++r) {
      int i = i0 + r + 8 * hi;
      int n = nt * 16 + lo;
      float v = acc[nt][r];
      h[(size_t)i * DOUT + n] = v;
      htp[(size_t)n * NN + i] = (__bf16)v;
    }
}

// ---------------- a2 = h @ w_a2 ----------------
__global__ void k_a2(const float* __restrict__ h, const float* __restrict__ wa2,
                     float* __restrict__ a2) {
  int i = blockIdx.x * blockDim.x + threadIdx.x;
  if (i >= NN) return;
  const float4* hr = (const float4*)(h + (size_t)i * DOUT);
  const float4* wv = (const float4*)wa2;
  float s = 0.0f;
#pragma unroll 8
  for (int t = 0; t < 32; ++t) {
    float4 a = hr[t], b = wv[t];
    s += a.x * b.x + a.y * b.y + a.z * b.z + a.w * b.w;
  }
  a2[i] = s;
}

// ---------------- global max of a2 ----------------
__global__ void k_max(const float* __restrict__ a2, float* __restrict__ Mout) {
  __shared__ float s[256];
  float m = -3.0e38f;
  for (int i = threadIdx.x; i < NN; i += 256) m = fmaxf(m, a2[i]);
  s[threadIdx.x] = m;
  __syncthreads();
  for (int off = 128; off > 0; off >>= 1) {
    if ((int)threadIdx.x < off) s[threadIdx.x] = fmaxf(s[threadIdx.x], s[threadIdx.x + off]);
    __syncthreads();
  }
  if (threadIdx.x == 0) Mout[0] = s[0];
}

// ---------------- e[j] = exp(a2[j] - M) as bf16 ----------------
__global__ void k_exp(const float* __restrict__ a2, const float* __restrict__ Mout,
                      __bf16* __restrict__ ebf) {
  int i = blockIdx.x * blockDim.x + threadIdx.x;
  if (i < NN) ebf[i] = (__bf16)__expf(a2[i] - Mout[0]);
}

// ---------------- TDM: stage ht_pad[0:144][k0:k0+32] into LDS ----------------
// 2D descriptor: data_size=2B, tile 32x144, tensor row stride 8192 elems.
// LDS padding: +2 dwords after every 16 dwords -> 72B row stride (bank-safe).
__device__ __forceinline__ void tdm_load_tile(const __bf16* gsrc, unsigned lds_off) {
  unsigned long long ga = (unsigned long long)(uintptr_t)gsrc;
  u32x4 g0;
  g0[0] = 1u;                                     // count=1, user mode
  g0[1] = lds_off;                                // lds_addr (bytes)
  g0[2] = (unsigned)(ga & 0xffffffffu);           // global_addr[31:0]
  g0[3] = (unsigned)((ga >> 32) & 0x01ffffffu)    // global_addr[56:32]
        | (2u << 30);                             // type=2 (image)
  i32x8 g1;
  g1[0] = (1 << 16)                               // data_size = 2 bytes
        | (1 << 20)                               // pad_enable
        | (3 << 22)                               // pad_interval: 16 dwords
        | (1 << 25);                              // pad_amount: 2 dwords
  g1[1] = (int)(8192u << 16);                     // tensor_dim0 = 8192 (lo16)
  g1[2] = (int)(144u << 16);                      // dim0 hi=0 | tensor_dim1=144
  g1[3] = (int)(32u << 16);                       // dim1 hi=0 | tile_dim0=32
  g1[4] = (int)(144u);                            // tile_dim1=144, tile_dim2=0
  g1[5] = (int)(8192u);                           // tensor_dim0_stride (lo32)
  g1[6] = 0;
  g1[7] = 0;
  // clang-23 / therock-10.0 form: 6 args (extra int32x8 group before cpol)
  __builtin_amdgcn_tensor_load_to_lds(g0, g1, (i32x4)0, (i32x4)0, (i32x8)0, 0);
}

// ---------------- main GEMM: partial[seg] = P_seg @ ht_pad ----------------
// Block = 8 waves, all on the SAME K-segment (8 distinct row strips) so the
// whole block shares one TDM-staged B tile stream (double-buffered in LDS).
__global__ void k_attn(const int* __restrict__ adj, const __bf16* __restrict__ ebf,
                       const __bf16* __restrict__ htp, float* __restrict__ partial) {
  __shared__ __bf16 tile[2][HTR * TROW];          // 2 x 10368 B

  const int lane  = threadIdx.x & 31;
  const int widx  = threadIdx.x >> 5;             // wave in block, 0..7
  const int seg   = blockIdx.x & (KSEG - 1);
  const int strip = (blockIdx.x >> 2) * 8 + widx; // 0..511
  const int lo = lane & 15, hi = lane >> 4, hi8 = hi * 8;
  const int i0 = strip * 16;
  const int kbeg = seg * KSEGL, kend = kbeg + KSEGL;
  const bool lead = (widx == 0);

  v8f acc[9];
#pragma unroll
  for (int t = 0; t < 9; ++t)
#pragma unroll
    for (int e = 0; e < 8; ++e) acc[t][e] = 0.0f;

  const int* arow = adj + (size_t)(i0 + lo) * NN;

  // -- prologue: stage first B tile, preload first adj/e chunk --
  if (lead) {
    tdm_load_tile(htp + kbeg, (unsigned)(uintptr_t)(&tile[0][0]));
    __builtin_amdgcn_s_wait_tensorcnt(0);
  }
  int4 q[4];
  v8bf ev2[2];
  q[0] = *(const int4*)(arow + kbeg + hi8);
  q[1] = *(const int4*)(arow + kbeg + hi8 + 4);
  q[2] = *(const int4*)(arow + kbeg + 16 + hi8);
  q[3] = *(const int4*)(arow + kbeg + 16 + hi8 + 4);
  ev2[0] = *(const v8bf*)(ebf + kbeg + hi8);
  ev2[1] = *(const v8bf*)(ebf + kbeg + 16 + hi8);
  __syncthreads();

  for (int k0 = kbeg; k0 < kend; k0 += 32) {
    const int cur = (k0 >> 5) & 1;
    const bool more = (k0 + 32) < kend;

    // prefetch next B tile into the other LDS buffer (TDM, async)
    if (lead && more)
      tdm_load_tile(htp + (k0 + 32), (unsigned)(uintptr_t)(&tile[cur ^ 1][0]));

    // software-pipelined adj/e loads for next iteration
    int4 qn[4];
    v8bf en[2];
    if (more) {
      const int kn = k0 + 32;
      __builtin_prefetch(arow + kn + 512, 0, 0);
      qn[0] = *(const int4*)(arow + kn + hi8);
      qn[1] = *(const int4*)(arow + kn + hi8 + 4);
      qn[2] = *(const int4*)(arow + kn + 16 + hi8);
      qn[3] = *(const int4*)(arow + kn + 16 + hi8 + 4);
      en[0] = *(const v8bf*)(ebf + kn + hi8);
      en[1] = *(const v8bf*)(ebf + kn + 16 + hi8);
    }

    // build A = P tile (16x32 bf16) from current adj/e
    int av[16] __attribute__((aligned(16)));
    *(int4*)(av + 0) = q[0]; *(int4*)(av + 4) = q[1];
    *(int4*)(av + 8) = q[2]; *(int4*)(av + 12) = q[3];
    __bf16 evv[16] __attribute__((aligned(16)));
    *(v8bf*)(evv + 0) = ev2[0]; *(v8bf*)(evv + 8) = ev2[1];
    v16bf A;
#pragma unroll
    for (int e = 0; e < 16; ++e) A[e] = (av[e] > 0) ? evv[e] : (__bf16)0.0f;

    // gather all 9 B tiles from LDS (bank-conflict-free b64 reads), then
    // issue the 9 WMMAs back-to-back
    v16bf Bt[9];
#pragma unroll
    for (int nt = 0; nt < 9; ++nt) {
      const __bf16* bp = &tile[cur][(nt * 16 + lo) * TROW + hi * 8];
      v4bf b0 = *(const v4bf*)(bp);
      v4bf b1 = *(const v4bf*)(bp + 4);
      v4bf b2 = *(const v4bf*)(bp + 16);
      v4bf b3 = *(const v4bf*)(bp + 20);
#pragma unroll
      for (int e = 0; e < 4; ++e) {
        Bt[nt][e] = b0[e]; Bt[nt][4 + e] = b1[e];
        Bt[nt][8 + e] = b2[e]; Bt[nt][12 + e] = b3[e];
      }
    }
#pragma unroll
    for (int nt = 0; nt < 9; ++nt)
      acc[nt] = __builtin_amdgcn_wmma_f32_16x16x32_bf16(
          false, A, false, Bt[nt], (short)0, acc[nt], false, false);

    // rotate the pipeline
    if (more) {
      q[0] = qn[0]; q[1] = qn[1]; q[2] = qn[2]; q[3] = qn[3];
      ev2[0] = en[0]; ev2[1] = en[1];
    }
    if (lead) __builtin_amdgcn_s_wait_tensorcnt(0);
    __syncthreads();
  }

  float* pseg = partial + (size_t)seg * NN * ACOL;
#pragma unroll
  for (int nt = 0; nt < 9; ++nt)
#pragma unroll
    for (int r = 0; r < 8; ++r)
      pseg[(size_t)(i0 + r + 8 * hi) * ACOL + nt * 16 + lo] = acc[nt][r];
}

// ---------------- reduce segments + normalize by Z ----------------
__global__ void k_norm(const float* __restrict__ partial, float* __restrict__ out) {
  int idx = blockIdx.x * blockDim.x + threadIdx.x;   // 8192*128
  if (idx >= NN * DOUT) return;
  int i = idx >> 7, n = idx & 127;
  float s = 0.0f, z = 0.0f;
#pragma unroll
  for (int g = 0; g < KSEG; ++g) {
    const float* p = partial + ((size_t)g * NN + i) * ACOL;
    s += p[n];
    z += p[128];
  }
  out[idx] = s / (z + 1e-30f);
}

// ---------------- launcher ----------------
extern "C" void kernel_launch(void* const* d_in, const int* in_sizes, int n_in,
                              void* d_out, int out_size, void* d_ws, size_t ws_size,
                              hipStream_t stream) {
  const float* x   = (const float*)d_in[0];
  const int*   adj = (const int*)d_in[1];
  const float* Wfc = (const float*)d_in[2];
  // d_in[3] = w_a1: unused — a1[i] is constant per softmax row and cancels.
  const float* wa2 = (const float*)d_in[4];
  float* out = (float*)d_out;

  char* ws = (char*)d_ws;
  __bf16* xbf = (__bf16*)ws;              ws += (size_t)NN * DIN * 2;   // 8 MB
  __bf16* wt  = (__bf16*)ws;              ws += (size_t)DOUT * DIN * 2; // 128 KB
  float*  h   = (float*)ws;               ws += (size_t)NN * DOUT * 4;  // 4 MB
  float*  a2  = (float*)ws;               ws += (size_t)NN * 4;         // 32 KB
  float*  Mv  = (float*)ws;               ws += 256;
  __bf16* ebf = (__bf16*)ws;              ws += (size_t)NN * 2;         // 16 KB
  __bf16* htp = (__bf16*)ws;              ws += (size_t)HTR * NN * 2;   // 2.25 MB
  float*  prt = (float*)ws;               // KSEG * 8192 * 144 * 4 = 18 MB

  k_cvt_x     <<<1024,               256, 0, stream>>>(x, xbf, NN * DIN);
  k_cvt_w     <<<(DOUT * DIN) / 256, 256, 0, stream>>>(Wfc, wt);
  k_init_htpad<<<(16 * NN) / 256,    256, 0, stream>>>(htp);
  k_fc        <<<64,                 256, 0, stream>>>(xbf, wt, h, htp); // 512 waves
  k_a2        <<<NN / 256,           256, 0, stream>>>(h, wa2, a2);
  k_max       <<<1,                  256, 0, stream>>>(a2, Mv);
  k_exp       <<<NN / 256,           256, 0, stream>>>(a2, Mv, ebf);
  k_attn      <<<256,                256, 0, stream>>>(adj, ebf, htp, prt); // 2048 waves
  k_norm      <<<(NN * DOUT) / 256,  256, 0, stream>>>(prt, out);
}
